// TransformerBlock_48095043781142
// MI455X (gfx1250) — matrix operand order repaired and checked by reading, not compiled, not run on hardware
//
#include <hip/hip_runtime.h>
#include <cstdint>
#include <cstddef>

// ---------------------------------------------------------------------------
// Problem constants (match reference)
// ---------------------------------------------------------------------------
#define T_SEQ   2048
#define N_B     2
#define DM      1024
#define N_H     16
#define DH      64
#define DFF     4096
#define M_ROWS  (N_B * T_SEQ)   // 4096
#define EPSV    1e-5f

// ---------------------------------------------------------------------------
// CDNA5 WMMA types / helpers (wave32; v_wmma_f32_16x16x32_bf16)
// ---------------------------------------------------------------------------
typedef __attribute__((ext_vector_type(16))) __bf16 v16bf;
typedef __attribute__((ext_vector_type(8)))  __bf16 v8bf;
typedef __attribute__((ext_vector_type(8)))  float  v8f;
typedef __attribute__((ext_vector_type(4)))  unsigned int u32x4;
typedef __attribute__((ext_vector_type(8)))  int i32x8;
typedef __attribute__((ext_vector_type(4)))  int i32x4;

static __device__ __forceinline__ v8f wmma_bf16(v16bf a, v16bf b, v8f c) {
  // 8 args: (neg_a, A, neg_b, B, c_mod, C, reuse_a, reuse_b)
  return __builtin_amdgcn_wmma_f32_16x16x32_bf16(false, a, false, b,
                                                 (short)0, c, false, false);
}

// A-matrix 16x32 bf16 fragment from a row-major row pointer.
// lane m = lane&15, g = lane>>4:
//   elems 0..7  -> K = g*16 + e
//   elems 8..15 -> K = g*16 + 8 + (e-8)
static __device__ __forceinline__ v16bf load_a_frag(const __bf16* rowp, int g) {
  v8bf lo = *(const v8bf*)(rowp + g * 16);
  v8bf hi = *(const v8bf*)(rowp + g * 16 + 8);
  v16bf r;
#pragma unroll
  for (int e = 0; e < 8; ++e) { r[e] = lo[e]; r[e + 8] = hi[e]; }
  return r;
}

// ---------------------------------------------------------------------------
// Tensor Data Mover: DMA a 64-row x 32-col bf16 tile (row-major, leading dim
// ld_elems) from global memory into LDS at byte offset lds_off.
// D# built per CDNA5 ISA ch.8: group0 {count,lds,global,type}, group1
// {data_size=2B, dims, tile 32x64, dim0 stride}.  Tracked on TENSORcnt.
// This toolchain exposes the 6-arg builtin form (g0,g1,g2,g3,g4,cpol).
// ---------------------------------------------------------------------------
static __device__ __forceinline__ void tdm_load_tile_bf16(
    const __bf16* gsrc, unsigned int lds_off, int ld_elems) {
  unsigned long long ga = (unsigned long long)(uintptr_t)gsrc;
  u32x4 g0;
  g0[0] = 1u;                                           // count=1, user mode
  g0[1] = lds_off;                                      // lds_addr
  g0[2] = (unsigned int)ga;                             // global_addr[31:0]
  g0[3] = (unsigned int)((ga >> 32) & 0x01FFFFFFu)      // global_addr[56:32]
          | (2u << 30);                                 // type=2 ("image")
  unsigned td0 = (unsigned)ld_elems;                    // tensor_dim0 (>=32)
  unsigned td1 = 65536u;                                // tensor_dim1 (>=64)
  i32x8 g1;
  g1[0] = (int)(1u << 16);                              // data_size=1 (2B)
  g1[1] = (int)((td0 & 0xFFFFu) << 16);                 // tensor_dim0 lo16
  g1[2] = (int)(((td0 >> 16) & 0xFFFFu) | ((td1 & 0xFFFFu) << 16));
  g1[3] = (int)(((td1 >> 16) & 0xFFFFu) | (32u << 16)); // tile_dim0 = 32
  g1[4] = (int)64;                                      // tile_dim1=64, dim2=0
  g1[5] = (int)(unsigned)ld_elems;                      // dim0 stride lo32
  g1[6] = 0;                                            // stride hi, dim1 stride
  g1[7] = 0;
  i32x4 z4 = {0, 0, 0, 0};
  i32x8 z8 = {0, 0, 0, 0, 0, 0, 0, 0};
  __builtin_amdgcn_tensor_load_to_lds(g0, g1, z4, z4, z8, 0);
}

// ---------------------------------------------------------------------------
// fp32 -> bf16 convert (weights)
// ---------------------------------------------------------------------------
__global__ void cvt_bf16_kernel(const float* __restrict__ in,
                                __bf16* __restrict__ out, int n) {
  int i = blockIdx.x * blockDim.x + threadIdx.x;
  if (i < n) out[i] = (__bf16)in[i];
}

// ---------------------------------------------------------------------------
// RMSNorm: one row (D=1024) per 128-thread block, fp32 in -> bf16 out
// ---------------------------------------------------------------------------
__global__ __launch_bounds__(128) void rmsnorm_kernel(
    const float* __restrict__ x, const float* __restrict__ w,
    __bf16* __restrict__ out, int D) {
  int row = blockIdx.x;
  const float* xr = x + (size_t)row * D;
  float ss = 0.0f;
  for (int i = threadIdx.x; i < D; i += 128) { float v = xr[i]; ss += v * v; }
  __shared__ float red[128];
  red[threadIdx.x] = ss;
  __syncthreads();
#pragma unroll
  for (int s = 64; s > 0; s >>= 1) {
    if ((int)threadIdx.x < s) red[threadIdx.x] += red[threadIdx.x + s];
    __syncthreads();
  }
  float scale = rsqrtf(red[0] / (float)D + EPSV);
  for (int i = threadIdx.x; i < D; i += 128)
    out[(size_t)row * D + i] = (__bf16)(xr[i] * scale * w[i]);
}

// ---------------------------------------------------------------------------
// RoPE + head transpose: fp32 (B,T,H*Dh) -> bf16 (B,H,T,Dh)
// ---------------------------------------------------------------------------
__global__ void rope_kernel(const float* __restrict__ src,
                            __bf16* __restrict__ dst) {
  int idx = blockIdx.x * blockDim.x + threadIdx.x;  // B*H*T*32 = 2^21 items
  int d2 = idx & 31;
  int t  = (idx >> 5) & (T_SEQ - 1);
  int h  = (idx >> 16) & (N_H - 1);
  int b  = idx >> 20;
  float freq = __powf(10000.0f, -((float)(2 * d2)) / (float)DH);
  float ang  = (float)t * freq;
  float sn, cs;
  __sincosf(ang, &sn, &cs);
  size_t sbase = ((size_t)(b * T_SEQ + t)) * DM + h * DH + 2 * d2;
  float x1 = src[sbase], x2 = src[sbase + 1];
  size_t dbase = (((size_t)(b * N_H + h)) * T_SEQ + t) * DH + 2 * d2;
  dst[dbase]     = (__bf16)(x1 * cs - x2 * sn);
  dst[dbase + 1] = (__bf16)(x1 * sn + x2 * cs);
}

// V: fp32 (B,T,H*Dh) -> bf16 (B,H,Dh,T)  (so PV B-fragments load contiguous K)
__global__ void vtrans_kernel(const float* __restrict__ src,
                              __bf16* __restrict__ dst) {
  int idx = blockIdx.x * blockDim.x + threadIdx.x;  // B*H*T*64 = 2^22 items
  int d = idx & (DH - 1);
  int t = (idx >> 6) & (T_SEQ - 1);
  int h = (idx >> 17) & (N_H - 1);
  int b = idx >> 21;
  dst[(((size_t)(b * N_H + h)) * DH + d) * T_SEQ + t] =
      (__bf16)src[((size_t)(b * T_SEQ + t)) * DM + h * DH + d];
}

// ---------------------------------------------------------------------------
// WMMA GEMM: C[M,N] = A[M,K] * W[N,K]^T   (A,W bf16 row-major, f32 accumulate)
// Block: 4 waves, block tile 128x64.  W 64x32 tile staged in LDS by the
// Tensor Data Mover (wave 0 issues, double-buffered, TENSORcnt + barrier).
// Each wave: 32x64 tile = 2 A-frags x 4 LDS B-frags -> 8 WMMAs per k-step.
// MODE 0: out fp32 = acc
// MODE 1: out fp32 = acc + aux            (residual)
// MODE 2: out bf16 = silu(aux) * acc      (fused SwiGLU gate)
// ---------------------------------------------------------------------------
template <int MODE>
__global__ __launch_bounds__(128) void gemm_wmma_kernel(
    const __bf16* __restrict__ A, const __bf16* __restrict__ W,
    const float* __restrict__ aux, float* __restrict__ outf,
    __bf16* __restrict__ outb, int M, int N, int K) {
  __shared__ __bf16 wtile[2][64 * 32];  // double-buffered 4KB W tiles

  int lane = threadIdx.x & 31;
  int wv   = threadIdx.x >> 5;
  int g    = lane >> 4;
  int l15  = lane & 15;
  int mb = blockIdx.y * 128 + wv * 32;
  int nb = blockIdx.x * 64;

  const __bf16* arow0 = A + (size_t)(mb + l15) * K;
  const __bf16* arow1 = A + (size_t)(mb + 16 + l15) * K;
  const __bf16* wbase = W + (size_t)nb * K;   // tile origin (row nb, col 0)
  unsigned int lds_buf[2] = {
      (unsigned int)(uintptr_t)&wtile[0][0],
      (unsigned int)(uintptr_t)&wtile[1][0]};

  v8f zero = {};
  v8f acc[2][4];
#pragma unroll
  for (int m = 0; m < 2; ++m)
#pragma unroll
    for (int t = 0; t < 4; ++t) acc[m][t] = zero;

  if (wv == 0) tdm_load_tile_bf16(wbase, lds_buf[0], K);

  for (int kb = 0; kb < K; kb += 32) {
    int cur = (kb >> 5) & 1;
    if (wv == 0) {
      if (kb + 32 < K) {
        tdm_load_tile_bf16(wbase + kb + 32, lds_buf[cur ^ 1], K);
        __builtin_amdgcn_s_wait_tensorcnt(1);  // tile kb landed in LDS
      } else {
        __builtin_amdgcn_s_wait_tensorcnt(0);
      }
    }
    __syncthreads();  // release consumers (uniform trip count in block)

    v16bf a0 = load_a_frag(arow0 + kb, g);
    v16bf a1 = load_a_frag(arow1 + kb, g);
    const __bf16* wt = &wtile[cur][0];
#pragma unroll
    for (int t = 0; t < 4; ++t) {
      // B-frag (32x16): lane col n = t*16+l15, elems e -> K = g*16+e
      v16bf bf = *(const v16bf*)(wt + (t * 16 + l15) * 32 + g * 16);
      acc[0][t] = wmma_bf16(a0, bf, acc[0][t]);
      acc[1][t] = wmma_bf16(a1, bf, acc[1][t]);
    }
    __syncthreads();  // all reads done before TDM overwrites this buffer
  }

#pragma unroll
  for (int m = 0; m < 2; ++m) {
#pragma unroll
    for (int t = 0; t < 4; ++t) {
#pragma unroll
      for (int i = 0; i < 8; ++i) {
        int row = mb + m * 16 + g * 8 + i;  // C layout: VGPR i -> M = g*8+i
        int col = nb + t * 16 + l15;        // lane -> N
        size_t idx = (size_t)row * (size_t)N + col;
        float a = acc[m][t][i];
        if constexpr (MODE == 0) {
          outf[idx] = a;
        } else if constexpr (MODE == 1) {
          outf[idx] = a + aux[idx];
        } else {
          float hh = aux[idx];
          float sg = 1.0f / (1.0f + __expf(-hh));
          outb[idx] = (__bf16)(hh * sg * a);
        }
      }
    }
  }
}

// ---------------------------------------------------------------------------
// Flash-style causal attention.
// Q,K in bf16 (B,H,T,Dh); V in bf16 (B,H,Dh,T); out bf16 (B,T,H*Dh).
// One wave per 16-query tile; key blocks of 32; streaming softmax.
// ---------------------------------------------------------------------------
__global__ __launch_bounds__(128) void attn_kernel(
    const __bf16* __restrict__ Q, const __bf16* __restrict__ K,
    const __bf16* __restrict__ V, __bf16* __restrict__ out) {
  __shared__ float pbuf[4][16 * 32];  // per-wave P staging (C-layout -> A-layout)

  int lane = threadIdx.x & 31;
  int wv   = threadIdx.x >> 5;
  int g    = lane >> 4;
  int l15  = lane & 15;
  int qb   = blockIdx.x * 64 + wv * 16;
  int bh   = blockIdx.y;             // b*N_H + h
  int b    = bh >> 4;
  int h    = bh & (N_H - 1);
  const float scale = 0.125f;        // 1/sqrt(Dh)

  // Q fragments: rows qb..qb+15, d split into two 32-wide k-steps
  const __bf16* qrow = Q + ((size_t)bh * T_SEQ + qb + l15) * DH;
  v16bf qa0 = load_a_frag(qrow, g);
  v16bf qa1 = load_a_frag(qrow + 32, g);

  v8f zero = {};
  v8f o[4];
#pragma unroll
  for (int t = 0; t < 4; ++t) o[t] = zero;
  float mrow[8], lrow[8];
#pragma unroll
  for (int i = 0; i < 8; ++i) { mrow[i] = -1e30f; lrow[i] = 0.0f; }

  float* prow_w = &pbuf[wv][0];

  for (int kb = 0; kb < qb + 16; kb += 32) {
    // ---- S = Q * K^T for two 16-key tiles --------------------------------
    v8f s[2];
#pragma unroll
    for (int nt = 0; nt < 2; ++nt) {
      int key = kb + nt * 16 + l15;
      const __bf16* krow = K + ((size_t)bh * T_SEQ + key) * DH;
      v16bf kf0 = *(const v16bf*)(krow + g * 16);        // d = g*16+e
      v16bf kf1 = *(const v16bf*)(krow + 32 + g * 16);   // d = 32+g*16+e
      v8f c = zero;
      c = wmma_bf16(qa0, kf0, c);
      c = wmma_bf16(qa1, kf1, c);
      s[nt] = c;
    }

    // ---- scale + causal mask + streaming softmax update ------------------
    int key0 = kb + l15;
    int key1 = kb + 16 + l15;
    float p0[8], p1[8];
#pragma unroll
    for (int i = 0; i < 8; ++i) {
      int row = qb + g * 8 + i;
      float v0 = s[0][i] * scale;
      float v1 = s[1][i] * scale;
      if (key0 > row) v0 = -1e30f;
      if (key1 > row) v1 = -1e30f;
      float mx = fmaxf(v0, v1);
#pragma unroll
      for (int msk = 1; msk < 16; msk <<= 1)
        mx = fmaxf(mx, __shfl_xor(mx, msk, 32));   // half-wave row reduce
      mx = fmaxf(mx, mrow[i]);
      float alpha = __expf(mrow[i] - mx);
      p0[i] = __expf(v0 - mx);
      p1[i] = __expf(v1 - mx);
      float sum = p0[i] + p1[i];
#pragma unroll
      for (int msk = 1; msk < 16; msk <<= 1)
        sum += __shfl_xor(sum, msk, 32);
      lrow[i] = lrow[i] * alpha + sum;
      mrow[i] = mx;
#pragma unroll
      for (int t = 0; t < 4; ++t) o[t][i] *= alpha;
    }

    // ---- transpose P (C-layout) -> A-layout via LDS ----------------------
#pragma unroll
    for (int i = 0; i < 8; ++i) {
      prow_w[(g * 8 + i) * 32 + l15]      = p0[i];
      prow_w[(g * 8 + i) * 32 + 16 + l15] = p1[i];
    }
    asm volatile("s_wait_dscnt 0" ::: "memory");   // in-wave DS WAR/RAW fence

    const float* prd = &pbuf[wv][l15 * 32];
    v16bf pf;
#pragma unroll
    for (int e = 0; e < 8; ++e) {
      pf[e]     = (__bf16)prd[g * 16 + e];         // key = g*16+e
      pf[e + 8] = (__bf16)prd[g * 16 + 8 + e];     // key = g*16+8+e
    }

    // ---- O += P * V ------------------------------------------------------
#pragma unroll
    for (int t = 0; t < 4; ++t) {
      int d = t * 16 + l15;
      const __bf16* vrow = V + ((size_t)bh * DH + d) * T_SEQ + kb + g * 16;
      v16bf vf = *(const v16bf*)vrow;              // keys g*16+e contiguous
      o[t] = wmma_bf16(pf, vf, o[t]);
    }
  }

  // ---- normalize + store to (B,T,H*Dh) -----------------------------------
#pragma unroll
  for (int t = 0; t < 4; ++t) {
#pragma unroll
    for (int i = 0; i < 8; ++i) {
      float val = o[t][i] / lrow[i];
      int row = qb + g * 8 + i;
      out[((size_t)(b * T_SEQ) + row) * DM + h * DH + t * 16 + l15] = (__bf16)val;
    }
  }
}

// ---------------------------------------------------------------------------
// Host-side orchestration
// ---------------------------------------------------------------------------
extern "C" void kernel_launch(void* const* d_in, const int* in_sizes, int n_in,
                              void* d_out, int out_size, void* d_ws,
                              size_t ws_size, hipStream_t stream) {
  const float* x_in  = (const float*)d_in[0];
  const float* rms1w = (const float*)d_in[1];
  const float* rms2w = (const float*)d_in[2];
  const float* q_w   = (const float*)d_in[3];
  const float* k_w   = (const float*)d_in[4];
  const float* v_w   = (const float*)d_in[5];
  const float* o_w   = (const float*)d_in[6];
  const float* w1    = (const float*)d_in[7];
  const float* w2    = (const float*)d_in[8];
  const float* w3    = (const float*)d_in[9];
  float* outp = (float*)d_out;

  // carve workspace
  char* p = (char*)d_ws;
  auto carve = [&](size_t bytes) -> void* {
    void* r = (void*)p;
    p += (bytes + 255) & ~(size_t)255;
    return r;
  };
  const size_t DD  = (size_t)DM * DM;      // 1M
  const size_t FD  = (size_t)DFF * DM;     // 4M
  const size_t MD  = (size_t)M_ROWS * DM;  // 4M
  const size_t MF  = (size_t)M_ROWS * DFF; // 16M
  __bf16* wq_b = (__bf16*)carve(DD * 2);
  __bf16* wk_b = (__bf16*)carve(DD * 2);
  __bf16* wv_b = (__bf16*)carve(DD * 2);
  __bf16* wo_b = (__bf16*)carve(DD * 2);
  __bf16* w1_b = (__bf16*)carve(FD * 2);
  __bf16* w3_b = (__bf16*)carve(FD * 2);
  __bf16* w2_b = (__bf16*)carve(FD * 2);
  __bf16* xn   = (__bf16*)carve(MD * 2);   // rmsnorm output (reused)
  float*  qf   = (float*)carve(MD * 4);
  float*  kf   = (float*)carve(MD * 4);
  float*  vf   = (float*)carve(MD * 4);
  __bf16* qr   = (__bf16*)carve(MD * 2);   // (B,H,T,Dh)
  __bf16* kr   = (__bf16*)carve(MD * 2);
  __bf16* vt   = (__bf16*)carve(MD * 2);   // (B,H,Dh,T)
  __bf16* attn = (__bf16*)carve(MD * 2);
  float*  y    = (float*)carve(MD * 4);    // residual stream after MHA
  float*  hbuf = (float*)carve(MF * 4);    // FFN h (pre-SiLU)
  __bf16* hg   = (__bf16*)carve(MF * 2);   // silu(h)*g

  // 1) weights -> bf16
  cvt_bf16_kernel<<<(int)(DD / 256), 256, 0, stream>>>(q_w, wq_b, (int)DD);
  cvt_bf16_kernel<<<(int)(DD / 256), 256, 0, stream>>>(k_w, wk_b, (int)DD);
  cvt_bf16_kernel<<<(int)(DD / 256), 256, 0, stream>>>(v_w, wv_b, (int)DD);
  cvt_bf16_kernel<<<(int)(DD / 256), 256, 0, stream>>>(o_w, wo_b, (int)DD);
  cvt_bf16_kernel<<<(int)(FD / 256), 256, 0, stream>>>(w1, w1_b, (int)FD);
  cvt_bf16_kernel<<<(int)(FD / 256), 256, 0, stream>>>(w3, w3_b, (int)FD);
  cvt_bf16_kernel<<<(int)(FD / 256), 256, 0, stream>>>(w2, w2_b, (int)FD);

  // 2) rmsnorm1
  rmsnorm_kernel<<<M_ROWS, 128, 0, stream>>>(x_in, rms1w, xn, DM);

  // 3) Q/K/V projections (M=4096, N=1024, K=1024), block tile 128x64
  dim3 g1(DM / 64, M_ROWS / 128);
  gemm_wmma_kernel<0><<<g1, 128, 0, stream>>>(xn, wq_b, nullptr, qf, nullptr,
                                              M_ROWS, DM, DM);
  gemm_wmma_kernel<0><<<g1, 128, 0, stream>>>(xn, wk_b, nullptr, kf, nullptr,
                                              M_ROWS, DM, DM);
  gemm_wmma_kernel<0><<<g1, 128, 0, stream>>>(xn, wv_b, nullptr, vf, nullptr,
                                              M_ROWS, DM, DM);

  // 4) RoPE + layout transforms
  rope_kernel<<<(N_B * N_H * T_SEQ * 32) / 256, 256, 0, stream>>>(qf, qr);
  rope_kernel<<<(N_B * N_H * T_SEQ * 32) / 256, 256, 0, stream>>>(kf, kr);
  vtrans_kernel<<<(N_B * N_H * T_SEQ * DH) / 256, 256, 0, stream>>>(vf, vt);

  // 5) causal attention
  dim3 ga(T_SEQ / 64, N_B * N_H);
  attn_kernel<<<ga, 128, 0, stream>>>(qr, kr, vt, attn);

  // 6) output projection + residual -> y
  gemm_wmma_kernel<1><<<g1, 128, 0, stream>>>(attn, wo_b, x_in, y, nullptr,
                                              M_ROWS, DM, DM);

  // 7) rmsnorm2 (reuse xn)
  rmsnorm_kernel<<<M_ROWS, 128, 0, stream>>>(y, rms2w, xn, DM);

  // 8) FFN: h = xn*w1^T ; hg = silu(h) * (xn*w3^T) ; out = y + hg*w2^T
  dim3 g2(DFF / 64, M_ROWS / 128);
  gemm_wmma_kernel<0><<<g2, 128, 0, stream>>>(xn, w1_b, nullptr, hbuf, nullptr,
                                              M_ROWS, DFF, DM);
  gemm_wmma_kernel<2><<<g2, 128, 0, stream>>>(xn, w3_b, hbuf, nullptr, hg,
                                              M_ROWS, DFF, DM);
  gemm_wmma_kernel<1><<<g1, 128, 0, stream>>>(hg, w2_b, y, outp, nullptr,
                                              M_ROWS, DM, DFF);
}
